// MVP_ori_9534827397532
// MI455X (gfx1250) — matrix operand order repaired and checked by reading, not compiled, not run on hardware
//
#include <hip/hip_runtime.h>
#include <hip/hip_bf16.h>
#include <math.h>

// ---------- problem geometry (static, matches reference) ----------
#define BG    128        // graphs
#define NN    512        // nodes per graph
#define NTOT  (BG*NN)    // 65536
#define DEGC  16
#define ETOT  (BG*NN*DEGC) // 1048576
#define GH    128
#define CID   64
#define CE    64
#define HAZ   256

#define WPACK_ELEMS (8*4*32*16)   // [tN][kk][lane][16] bf16 fragments of W = 32KB

typedef __attribute__((ext_vector_type(16))) __bf16 v16bf;
typedef __attribute__((ext_vector_type(8)))  float  v8f;
typedef __attribute__((ext_vector_type(4)))  unsigned int u32x4;
typedef __attribute__((ext_vector_type(8)))  int i32x8;
typedef __attribute__((ext_vector_type(4)))  int i32x4;

#if defined(__has_builtin)
#if __has_builtin(__builtin_amdgcn_tensor_load_to_lds) && __has_builtin(__builtin_amdgcn_s_wait_tensorcnt)
#define HAVE_TDM 1
#endif
#endif

__device__ __forceinline__ __bf16 f2bf(float f) {
  union { float f; unsigned u; } a; a.f = f;
  unsigned r = a.u + 0x7FFFu + ((a.u >> 16) & 1u);   // round-to-nearest-even
  union { unsigned short s; __bf16 b; } c;
  c.s = (unsigned short)(r >> 16);
  return c.b;
}

// ---------- init: xcur = x, mask = 1 ----------
__global__ void init_kernel(const float* __restrict__ x, float* __restrict__ xcur,
                            float* __restrict__ mask) {
  long long i = (long long)blockIdx.x * blockDim.x + threadIdx.x;
  if (i < (long long)NTOT * GH) xcur[i] = x[i];
  if (i < NTOT) mask[i] = 1.0f;
}

// ---------- pack A: f32 row-major -> bf16 row-major (convert once per stage) ----------
__global__ void packA_kernel(const float* __restrict__ X, __bf16* __restrict__ Abf) {
  long long i = (long long)blockIdx.x * blockDim.x + threadIdx.x;
  if (i < (long long)NTOT * GH) Abf[i] = f2bf(X[i]);
}

// ---------- pack W: f32 [128x128] -> fragment-major bf16 [tN][kk][lane][16] ----------
__global__ void packW_kernel(const float* __restrict__ W, __bf16* __restrict__ Wpack) {
  int i = blockIdx.x * blockDim.x + threadIdx.x;
  if (i >= WPACK_ELEMS) return;
  const int j    = i & 15;
  const int lane = (i >> 4) & 31;
  const int kk   = (i >> 9) & 3;
  const int tN   = i >> 11;
  const int krow = kk * 32 + (lane >> 4) * 16 + j;
  const int col  = tN * 16 + (lane & 15);
  Wpack[i] = f2bf(W[krow * GH + col]);
}

// ---------- GEMM  C[M x 128] = A[M x 128] * W[128 x 128]  via WMMA bf16 ----------
// One workgroup (8 waves) per 128-row strip. Packed W staged in LDS once via the
// Tensor Data Mover (one 32KB 1-D tile descriptor, TENSORcnt-tracked); each wave
// keeps its 4 A-fragments in registers and sweeps all 8 N-tiles -> 32 WMMA/wave.
__global__ void wmma_gemm_kernel(const __bf16* __restrict__ Abf,
                                 const __bf16* __restrict__ Wpack,
                                 float* __restrict__ C, int M) {
  __shared__ __bf16 wsm[WPACK_ELEMS];                 // 32 KB
  const int tid = threadIdx.x;

#ifdef HAVE_TDM
  if ((tid >> 5) == 0) {
    // Tensor DMA descriptor (D#), ISA 8.3/8.4: 1-D tile, 8192 elems x 4 bytes.
    const unsigned long long ga = (unsigned long long)Wpack;   // global addr of tile
    const unsigned lds_addr = (unsigned)(unsigned long long)(void*)wsm; // LDS aperture low 32b = LDS offset
    u32x4 g0;
    g0[0] = 1u;                                        // count=1, user descriptor
    g0[1] = lds_addr;                                  // lds_addr[31:0]
    g0[2] = (unsigned)(ga & 0xFFFFFFFFu);              // global_addr[31:0]
    g0[3] = (unsigned)((ga >> 32) & 0x1FFFFFFu) | (2u << 30);  // global_addr[56:32] | type=2
    i32x8 g1;
    g1[0] = (2 << 16);                                 // workgroup_mask=0, data_size=2 (4 bytes)
    g1[1] = (int)((8192u & 0xFFFFu) << 16);            // tensor_dim0[15:0] = 8192
    g1[2] = 0;                                         // tensor_dim0 hi / tensor_dim1 lo
    g1[3] = (int)(8192u << 16);                        // tile_dim0 = 8192
    g1[4] = 0;                                         // tile_dim1=0 (unused), tile_dim2=0
    g1[5] = 8192;                                      // tensor_dim0_stride lo
    g1[6] = 0;
    g1[7] = 0;
    i32x4 z4 = {0, 0, 0, 0};
#if __clang_major__ >= 23
    i32x8 z8 = {0, 0, 0, 0, 0, 0, 0, 0};
    __builtin_amdgcn_tensor_load_to_lds(g0, g1, z4, z4, z8, 0);
#else
    __builtin_amdgcn_tensor_load_to_lds(g0, g1, z4, z4, 0);
#endif
    __builtin_amdgcn_s_wait_tensorcnt(0);
  }
#else
  {   // fallback: cooperative 16B-vector stage of packed W into LDS
    const uint4* gs = (const uint4*)Wpack;
    uint4* gd = (uint4*)wsm;
    for (int i = tid; i < WPACK_ELEMS / 8; i += blockDim.x) gd[i] = gs[i];
  }
#endif
  __syncthreads();

  const int wave = tid >> 5, lane = tid & 31;
  const int tM   = blockIdx.x * 8 + wave;
  if (tM * 16 >= M) return;
  const int half = lane >> 4;
  const int row  = tM * 16 + (lane & 15);

  // A fragments for all 4 k-steps: two contiguous 16B chunks each
  union Frag { v16bf v; uint4 q[2]; };
  Frag afrag[4];
  const uint4* aq = (const uint4*)(Abf + (long long)row * GH);  // 16 uint4 per row
  #pragma unroll
  for (int kk = 0; kk < 4; ++kk) {
    afrag[kk].q[0] = aq[kk * 4 + 2 * half];        // K = kk*32 + 16*half + {0..7}
    afrag[kk].q[1] = aq[kk * 4 + 2 * half + 1];    // K = kk*32 + 16*half + 8 + {0..7}
  }

  const uint4* bq = (const uint4*)wsm;
  #pragma unroll
  for (int tN = 0; tN < 8; ++tN) {
    v8f acc = {};
    #pragma unroll
    for (int kk = 0; kk < 4; ++kk) {
      Frag bf;
      const int base = ((tN * 4 + kk) * 32 + lane) * 2;  // uint4 index
      bf.q[0] = bq[base];
      bf.q[1] = bq[base + 1];
      acc = __builtin_amdgcn_wmma_f32_16x16x32_bf16(false, afrag[kk].v, false, bf.v,
                                                    (short)0, acc, false, false);
    }
    float* cp = C + (long long)(tM * 16 + half * 8) * GH + tN * 16 + (lane & 15);
    #pragma unroll
    for (int r = 0; r < 8; ++r) cp[(long long)r * GH] = acc[r];
  }
}

// ---------- degree: deg = mask (self loop) + sum over alive edges ----------
__global__ void deg_init_kernel(const float* __restrict__ mask, float* __restrict__ deg) {
  int i = blockIdx.x * blockDim.x + threadIdx.x;
  if (i < NTOT) deg[i] = mask[i];
}
__global__ void deg_edge_kernel(const int* __restrict__ src, const int* __restrict__ dst,
                                const float* __restrict__ mask, float* __restrict__ deg) {
  int e = blockIdx.x * blockDim.x + threadIdx.x;
  if (e >= ETOT) return;
  float w = mask[src[e]] * mask[dst[e]];
  if (w > 0.0f) atomicAdd(&deg[dst[e]], w);
}
__global__ void dinv_kernel(const float* __restrict__ deg, float* __restrict__ dinv) {
  int i = blockIdx.x * blockDim.x + threadIdx.x;
  if (i < NTOT) dinv[i] = rsqrtf(fmaxf(deg[i], 1.0f));
}

// ---------- aggregation: one workgroup per graph, LDS tile 512 nodes x 32 dims ----------
// out = relu((segsum(norm*h[src]) + h*mask*dinv^2 + bias) * mask)
__global__ void agg_kernel(const float* __restrict__ h,
                           const int* __restrict__ src, const int* __restrict__ dst,
                           const float* __restrict__ mask, const float* __restrict__ dinv,
                           const float* __restrict__ bias, float* __restrict__ xout) {
  __shared__ float acc[NN * 32];                      // 64 KB of the 320 KB WGP LDS
  const int g = blockIdx.x, tid = threadIdx.x;        // 256 threads
  const int ebase = g * (NN * DEGC), nbase = g * NN;
  for (int t = 0; t < GH / 32; ++t) {
    for (int i = tid; i < NN * 32; i += blockDim.x) acc[i] = 0.0f;
    __syncthreads();
    for (int e = tid; e < NN * DEGC; e += blockDim.x) {
      {   // prefetch next iteration's gathered row (global_prefetch_b8)
        const int e2 = e + blockDim.x;
        if (e2 < NN * DEGC)
          __builtin_prefetch(h + (long long)src[ebase + e2] * GH + t * 32, 0, 1);
      }
      const int s = src[ebase + e], d = dst[ebase + e];
      const float w = mask[s] * mask[d];
      if (w > 0.0f) {
        const float nrm = w * dinv[s] * dinv[d];
        const float* hs = h + (long long)s * GH + t * 32;
        const int dl = (d - nbase) * 32;
        #pragma unroll
        for (int dd = 0; dd < 32; ++dd) atomicAdd(&acc[dl + dd], nrm * hs[dd]);  // ds_add_f32
      }
    }
    __syncthreads();
    for (int i = tid; i < NN * 32; i += blockDim.x) {
      const int node = i >> 5, dd = i & 31;
      const int gi = nbase + node;
      const float m = mask[gi], di = dinv[gi];
      const float self = h[(long long)gi * GH + t * 32 + dd] * m * di * di;
      const float v = (acc[i] + self + bias[t * 32 + dd]) * m;
      xout[(long long)gi * GH + t * 32 + dd] = fmaxf(v, 0.0f);
    }
    __syncthreads();
  }
}

// ---------- pooling: score -> per-graph bitonic top-k -> gate + new mask ----------
__global__ void pool_kernel(float* __restrict__ x, float* __restrict__ mask,
                            const float* __restrict__ p, int k) {
  __shared__ float key[NN];
  __shared__ int   sidx[NN];
  __shared__ float sc[NN];
  __shared__ float nm[NN];
  const int g = blockIdx.x, tid = threadIdx.x;        // 256 threads
  const int nbase = g * NN;

  float pn = 0.0f;
  for (int d = 0; d < GH; ++d) pn += p[d] * p[d];
  const float pinv = rsqrtf(fmaxf(pn, 1e-30f));

  for (int i = tid; i < NN; i += blockDim.x) {
    const float* xr = x + (long long)(nbase + i) * GH;
    float dot = 0.0f;
    for (int d = 0; d < GH; ++d) dot += xr[d] * p[d];
    const float s = 1.0f / (1.0f + __expf(-dot * pinv));
    sc[i]  = s;
    key[i] = (mask[nbase + i] > 0.0f) ? s : -3.4e38f;
    sidx[i] = i;
    nm[i] = 0.0f;
  }
  __syncthreads();

  // bitonic sort: descending key, ascending index on ties (matches lax.top_k)
  for (int k2 = 2; k2 <= NN; k2 <<= 1) {
    for (int j = k2 >> 1; j > 0; j >>= 1) {
      for (int t = tid; t < NN; t += blockDim.x) {
        const int ixj = t ^ j;
        if (ixj > t) {
          const bool up = ((t & k2) == 0);
          const float ka = key[t], kb = key[ixj];
          const int ia = sidx[t], ib = sidx[ixj];
          const bool before = (ka > kb) || (ka == kb && ia < ib);
          if (up ? !before : before) {
            key[t] = kb; key[ixj] = ka; sidx[t] = ib; sidx[ixj] = ia;
          }
        }
      }
      __syncthreads();
    }
  }
  for (int t = tid; t < NN; t += blockDim.x) if (t < k) nm[sidx[t]] = 1.0f;
  __syncthreads();

  for (int i = tid; i < NN; i += blockDim.x) {
    const float m = nm[i];
    mask[nbase + i] = m;
    const float scale = sc[i] * m;
    float* xr = x + (long long)(nbase + i) * GH;
    for (int d = 0; d < GH; ++d) xr[d] *= scale;
  }
}

// ---------- readout: hg (or +=) concat(relu(max), relu(mean)) ----------
__global__ void readout_kernel(const float* __restrict__ x, const float* __restrict__ mask,
                               float* __restrict__ hg, int accumulate) {
  const int g = blockIdx.x, d = threadIdx.x;          // 128 threads, one per feature
  const int nbase = g * NN;
  float mx = -1e30f, sm = 0.0f, cnt = 0.0f;
  for (int i = 0; i < NN; ++i) {
    const float m = mask[nbase + i];
    const float v = x[(long long)(nbase + i) * GH + d];
    if (m > 0.0f && v > mx) mx = v;
    sm += v * m;
    cnt += m;
  }
  const float mean = sm / fmaxf(cnt, 1.0f);
  const float rmx = fmaxf(mx, 0.0f), rmean = fmaxf(mean, 0.0f);
  if (accumulate) { hg[g * (2 * GH) + d] += rmx; hg[g * (2 * GH) + GH + d] += rmean; }
  else            { hg[g * (2 * GH) + d]  = rmx; hg[g * (2 * GH) + GH + d]  = rmean; }
}

// ---------- tail MLP ----------
__global__ void embed_kernel(const float* __restrict__ inp_c, const float* __restrict__ We,
                             float* __restrict__ embed) {
  const int b = blockIdx.x, j = threadIdx.x;          // 64 threads
  float s = 0.0f;
  for (int d = 0; d < CID; ++d) s += inp_c[b * CID + d] * We[d * CE + j];
  embed[b * CE + j] = fmaxf(s, 0.0f);
}
__global__ void fc1_kernel(const float* __restrict__ embed, const float* __restrict__ hg,
                           const float* __restrict__ Wa, const float* __restrict__ ba,
                           float* __restrict__ ha) {
  const int b = blockIdx.x, j = threadIdx.x;          // 256 threads
  float s = ba[j];
  for (int d = 0; d < CE; ++d)       s += embed[b * CE + d] * Wa[d * HAZ + j];
  for (int d = 0; d < 2 * GH; ++d)   s += hg[b * (2 * GH) + d] * Wa[(CE + d) * HAZ + j];
  ha[b * HAZ + j] = fmaxf(s, 0.0f);
}
__global__ void fc2_kernel(const float* __restrict__ ha, const float* __restrict__ Wb,
                           const float* __restrict__ bb, float* __restrict__ hb) {
  const int b = blockIdx.x, j = threadIdx.x;          // 128 threads
  float s = bb[j];
  for (int d = 0; d < HAZ; ++d) s += ha[b * HAZ + d] * Wb[d * (HAZ / 2) + j];
  hb[b * (HAZ / 2) + j] = fmaxf(s, 0.0f);
}
__global__ void out_kernel(const float* __restrict__ hb, const float* __restrict__ Wc,
                           float* __restrict__ out) {
  const int b = threadIdx.x;                          // 128 threads, one block
  if (b >= BG) return;
  float s = 0.0f;
  for (int d = 0; d < HAZ / 2; ++d) s += hb[b * (HAZ / 2) + d] * Wc[d];
  out[b] = s;
}

extern "C" void kernel_launch(void* const* d_in, const int* in_sizes, int n_in,
                              void* d_out, int out_size, void* d_ws, size_t ws_size,
                              hipStream_t stream) {
  const float* x     = (const float*)d_in[0];
  const float* inp_c = (const float*)d_in[1];
  const int*   src   = (const int*)d_in[2];
  const int*   dst   = src + ETOT;                    // edge_index = [src; dst]
  const float* W[3]  = { (const float*)d_in[4], (const float*)d_in[6], (const float*)d_in[8] };
  const float* bv[3] = { (const float*)d_in[5], (const float*)d_in[7], (const float*)d_in[9] };
  const float* pv[3] = { (const float*)d_in[10], (const float*)d_in[11], (const float*)d_in[12] };
  const float* We = (const float*)d_in[13];
  const float* Wa = (const float*)d_in[14];
  const float* ba = (const float*)d_in[15];
  const float* Wb = (const float*)d_in[16];
  const float* bb = (const float*)d_in[17];
  const float* Wc = (const float*)d_in[18];
  float* out = (float*)d_out;

  // workspace layout
  char* ws = (char*)d_ws;
  const size_t szX = (size_t)NTOT * GH * sizeof(float);
  float*  xcur  = (float*)ws;                ws += szX;
  float*  hbuf  = (float*)ws;                ws += szX;
  __bf16* Abf   = (__bf16*)ws;               ws += (size_t)NTOT * GH * sizeof(__bf16);
  __bf16* Wpack = (__bf16*)ws;               ws += (size_t)WPACK_ELEMS * sizeof(__bf16);
  float*  mask  = (float*)ws;                ws += (size_t)NTOT * sizeof(float);
  float*  deg   = (float*)ws;                ws += (size_t)NTOT * sizeof(float);
  float*  dinv  = (float*)ws;                ws += (size_t)NTOT * sizeof(float);
  float*  hg    = (float*)ws;                ws += (size_t)BG * 2 * GH * sizeof(float);
  float*  embed = (float*)ws;                ws += (size_t)BG * CE * sizeof(float);
  float*  ha    = (float*)ws;                ws += (size_t)BG * HAZ * sizeof(float);
  float*  hb    = (float*)ws;                ws += (size_t)BG * (HAZ / 2) * sizeof(float);
  (void)ws_size; (void)n_in; (void)in_sizes; (void)out_size;

  const int Ks[3] = { 256, 128, 64 };

  init_kernel<<<(NTOT * GH + 255) / 256, 256, 0, stream>>>(x, xcur, mask);

  for (int s = 0; s < 3; ++s) {
    // pack operands to bf16 (convert once), then h = xcur @ W[s] via WMMA
    packA_kernel<<<(NTOT * GH + 255) / 256, 256, 0, stream>>>(xcur, Abf);
    packW_kernel<<<(WPACK_ELEMS + 255) / 256, 256, 0, stream>>>(W[s], Wpack);
    wmma_gemm_kernel<<<NTOT / 128, 256, 0, stream>>>(Abf, Wpack, hbuf, NTOT);
    // degree / norm
    deg_init_kernel<<<NTOT / 256, 256, 0, stream>>>(mask, deg);
    deg_edge_kernel<<<ETOT / 256, 256, 0, stream>>>(src, dst, mask, deg);
    dinv_kernel<<<NTOT / 256, 256, 0, stream>>>(deg, dinv);
    // edge aggregation + self loop + bias + relu  (LDS per-graph accumulation)
    agg_kernel<<<BG, 256, 0, stream>>>(hbuf, src, dst, mask, dinv, bv[s], xcur);
    // top-k pooling (score gate + new mask)
    pool_kernel<<<BG, 256, 0, stream>>>(xcur, mask, pv[s], Ks[s]);
    // readout accumulate into hg
    readout_kernel<<<BG, GH, 0, stream>>>(xcur, mask, hg, s == 0 ? 0 : 1);
  }

  // tail MLP
  embed_kernel<<<BG, CE, 0, stream>>>(inp_c, We, embed);
  fc1_kernel<<<BG, HAZ, 0, stream>>>(embed, hg, Wa, ba, ha);
  fc2_kernel<<<BG, HAZ / 2, 0, stream>>>(ha, Wb, bb, hb);
  out_kernel<<<1, 128, 0, stream>>>(hb, Wc, out);
}
